// TransformerEncoderCustomV5_59090160059079
// MI455X (gfx1250) — compile-verified, matrix operand order checked by reading, not run on hardware
//
#include <hip/hip_runtime.h>
#include <hip/hip_bf16.h>
#include <math.h>

// ---------------------------------------------------------------------------
// Types for CDNA5 WMMA / TDM (gfx1250, wave32)
// ---------------------------------------------------------------------------
typedef __bf16 bf16_t;
typedef __attribute__((ext_vector_type(8)))  bf16_t v8bf;
typedef __attribute__((ext_vector_type(16))) bf16_t v16bf;
typedef __attribute__((ext_vector_type(8)))  float  v8f;
typedef __attribute__((ext_vector_type(4)))  unsigned int v4u;
typedef __attribute__((ext_vector_type(4)))  int v4i;
typedef __attribute__((ext_vector_type(8)))  int v8i;

constexpr int kN  = 32;
constexpr int kT  = 512;
constexpr int kD  = 768;
constexpr int kH  = 12;
constexpr int kDH = 64;
constexpr int kFF = 3072;
constexpr int kL  = 6;
constexpr int kC  = 256;
constexpr int kM  = kN * kT;   // 16384 rows of the (N*T, D) activation matrix

#if defined(__has_builtin)
#if __has_builtin(__builtin_amdgcn_tensor_load_to_lds) && \
    __has_builtin(__builtin_amdgcn_s_wait_tensorcnt)
#define USE_TDM 1
#endif
#endif
#ifndef USE_TDM
#define USE_TDM 0
#endif

// ---------------------------------------------------------------------------
// TDM: issue a 2D tile load (bf16 elements) global -> LDS.
// D# layout per CDNA5 ISA ch.8: group0 = {count/type/addr}, group1 = dims.
// Row padding: tile_dim0 = 32 elems = 16 DWORDs (pad_interval code 3), pad
// amount = 4 DWORDs (code 3) -> LDS row stride 40 bf16 (conflict-free b128).
// ---------------------------------------------------------------------------
#if USE_TDM
__device__ __forceinline__ void tdm_load_2d(unsigned lds_addr, const void* gptr,
                                            unsigned tensor_d0, unsigned tensor_d1,
                                            unsigned tile_d0, unsigned tile_d1,
                                            unsigned long long stride0_elems) {
  unsigned long long ga = (unsigned long long)(size_t)gptr;
  v4u g0;
  g0[0] = 1u;                                              // count=1, user D#
  g0[1] = lds_addr;                                        // LDS byte address
  g0[2] = (unsigned)(ga & 0xffffffffull);                  // global_addr[31:0]
  g0[3] = (unsigned)((ga >> 32) & 0x1ffffffull) | (2u << 30);  // [56:32], type=2
  v8i g1;
  g1[0] = (int)((1u << 16)        // data_size = 2 bytes
              | (1u << 20)        // pad_enable
              | (3u << 22)        // pad_interval: 16 DWORDs
              | (3u << 25));      // pad_amount: 4 DWORDs
  g1[1] = (int)((tensor_d0 & 0xffffu) << 16);              // tensor_dim0[15:0]
  g1[2] = (int)((tensor_d0 >> 16) | ((tensor_d1 & 0xffffu) << 16));
  g1[3] = (int)((tensor_d1 >> 16) | (tile_d0 << 16));      // tile_dim0
  g1[4] = (int)(tile_d1 & 0xffffu);                        // tile_dim1 (tile_dim2=0)
  g1[5] = (int)(unsigned)(stride0_elems & 0xffffffffull);  // dim0 stride lo
  g1[6] = (int)(unsigned)((stride0_elems >> 32) & 0xffffull);
  g1[7] = 0;
  v4i gz = {0, 0, 0, 0};
#if defined(__clang_major__) && __clang_major__ >= 23
  v8i gz8 = {0, 0, 0, 0, 0, 0, 0, 0};
  __builtin_amdgcn_tensor_load_to_lds(g0, g1, gz, gz, gz8, 0);
#else
  __builtin_amdgcn_tensor_load_to_lds(g0, g1, gz, gz, 0);
#endif
}
#endif

// ---------------------------------------------------------------------------
// Transpose + fp32->bf16 weight conversion: dst[n*K + k] = (bf16)src[k*Nn + n]
// (batched over blockIdx.z matrices). 32x32 LDS tile, both sides coalesced.
// All GEMM B operands become row-major-in-(n,k), so GEMM staging needs no
// transpose and maps directly onto TDM 2D tile loads.
// ---------------------------------------------------------------------------
__global__ __launch_bounds__(256)
void cvtT_kernel(const float* __restrict__ src, bf16_t* __restrict__ dst,
                 int K, int Nn) {
  __shared__ float tile[32][33];
  const int mat = blockIdx.z;
  const float* s = src + (size_t)mat * K * Nn;
  bf16_t* d = dst + (size_t)mat * K * Nn;
  const int n0 = blockIdx.x * 32, k0 = blockIdx.y * 32;
  const int tx = threadIdx.x, ty = threadIdx.y;  // 32 x 8
#pragma unroll
  for (int j = 0; j < 32; j += 8)
    tile[ty + j][tx] = s[(size_t)(k0 + ty + j) * Nn + n0 + tx];
  __syncthreads();
#pragma unroll
  for (int j = 0; j < 32; j += 8)
    d[(size_t)(n0 + ty + j) * K + k0 + tx] = (bf16_t)tile[tx][ty + j];
}

// ---------------------------------------------------------------------------
// LayerNorm: fp32 in, bf16 out (feeds WMMA). One 256-thread block per row.
// ---------------------------------------------------------------------------
__global__ __launch_bounds__(256)
void ln_kernel(const float* __restrict__ x, const float* __restrict__ sc,
               const float* __restrict__ bi, bf16_t* __restrict__ y) {
  const int row = blockIdx.x, tid = threadIdx.x;
  const float* xr = x + (size_t)row * kD;
  float v[3];
  float s = 0.f, q = 0.f;
#pragma unroll
  for (int j = 0; j < 3; ++j) {
    v[j] = xr[tid + j * 256];
    s += v[j];
    q += v[j] * v[j];
  }
#pragma unroll
  for (int off = 16; off; off >>= 1) {
    s += __shfl_down(s, off, 32);
    q += __shfl_down(q, off, 32);
  }
  __shared__ float rs[8], rq[8];
  __shared__ float mu_s, rstd_s;
  if ((tid & 31) == 0) { rs[tid >> 5] = s; rq[tid >> 5] = q; }
  __syncthreads();
  if (tid == 0) {
    float S = 0.f, Q = 0.f;
    for (int i = 0; i < 8; ++i) { S += rs[i]; Q += rq[i]; }
    float mu  = S * (1.0f / (float)kD);
    float var = Q * (1.0f / (float)kD) - mu * mu;
    mu_s = mu;
    rstd_s = rsqrtf(var + 1e-5f);
  }
  __syncthreads();
  const float mu = mu_s, rstd = rstd_s;
  bf16_t* yr = y + (size_t)row * kD;
#pragma unroll
  for (int j = 0; j < 3; ++j) {
    int c = tid + j * 256;
    yr[c] = (bf16_t)((v[j] - mu) * rstd * sc[c] + bi[c]);
  }
}

// ---------------------------------------------------------------------------
// Tiled bf16 WMMA GEMM:  C[M,Nn] = A[M,K] * Bt[Nn,K]^T + bias
//   mode 0: store bf16 | mode 1: exact-erf GELU, store bf16
//   mode 2: fp32 residual add into Cres
// 256 threads = 8 waves; block tile 128x128; wave tile 32x64 (2x4 WMMA frags).
// K staged 32-wide. TDM path: double-buffered tensor_load_to_lds issued by
// wave 0, s_wait_tensorcnt + barrier; rows padded to 40 bf16 so fragment
// loads are two contiguous ds_load_b128 per lane (ISA 16-bit operand layout:
// K(e,half) = (e/8)*16 + half*8 + e%8).
// ---------------------------------------------------------------------------
#define BM 128
#define BN 128
#define BK 32
#define LDK (BK + 8)

__device__ __forceinline__ void compute_tile(const bf16_t (*Asb)[LDK],
                                             const bf16_t (*Bsb)[LDK],
                                             int wm, int wn, int half, int r,
                                             v8f acc[2][4]) {
  v16bf afr[2], bfr[4];
#pragma unroll
  for (int mi = 0; mi < 2; ++mi) {
    int row = wm * 32 + mi * 16 + r;
    v8bf lo = *(const v8bf*)(&Asb[row][half * 8]);
    v8bf hi = *(const v8bf*)(&Asb[row][16 + half * 8]);
    afr[mi] = __builtin_shufflevector(lo, hi, 0, 1, 2, 3, 4, 5, 6, 7,
                                      8, 9, 10, 11, 12, 13, 14, 15);
  }
#pragma unroll
  for (int ni = 0; ni < 4; ++ni) {
    int col = wn * 64 + ni * 16 + r;
    v8bf lo = *(const v8bf*)(&Bsb[col][half * 8]);
    v8bf hi = *(const v8bf*)(&Bsb[col][16 + half * 8]);
    bfr[ni] = __builtin_shufflevector(lo, hi, 0, 1, 2, 3, 4, 5, 6, 7,
                                      8, 9, 10, 11, 12, 13, 14, 15);
  }
#pragma unroll
  for (int mi = 0; mi < 2; ++mi)
#pragma unroll
    for (int ni = 0; ni < 4; ++ni)
      acc[mi][ni] = __builtin_amdgcn_wmma_f32_16x16x32_bf16(
          false, afr[mi], false, bfr[ni], (short)0, acc[mi][ni], false, false);
}

__global__ __launch_bounds__(256)
void gemm_bf16_kernel(const bf16_t* __restrict__ A, const bf16_t* __restrict__ Bt,
                      const float* __restrict__ bias, bf16_t* __restrict__ Cbf,
                      float* __restrict__ Cres, int M, int Nn, int K, int mode) {
  __shared__ __align__(16) bf16_t As[2][BM][LDK];
  __shared__ __align__(16) bf16_t Bs[2][BN][LDK];

  const int tid  = threadIdx.x;
  const int lane = tid & 31;
  const int wv   = tid >> 5;
  const int wm   = wv >> 1;          // 0..3 (M)
  const int wn   = wv & 1;           // 0..1 (N)
  const int half = lane >> 4;
  const int r    = lane & 15;
  const int m0   = blockIdx.y * BM;
  const int n0   = blockIdx.x * BN;

  v8f zero = {};
  v8f acc[2][4];
#pragma unroll
  for (int mi = 0; mi < 2; ++mi)
#pragma unroll
    for (int ni = 0; ni < 4; ++ni) acc[mi][ni] = zero;

  const int nk = K / BK;

#if USE_TDM
  if (wv == 0) {
    tdm_load_2d((unsigned)(size_t)&As[0][0][0], A + (size_t)m0 * K,
                (unsigned)K, (unsigned)M, BK, BM, (unsigned long long)K);
    tdm_load_2d((unsigned)(size_t)&Bs[0][0][0], Bt + (size_t)n0 * K,
                (unsigned)K, (unsigned)Nn, BK, BN, (unsigned long long)K);
  }
  int cur = 0;
  for (int it = 0; it < nk; ++it) {
    if (wv == 0) __builtin_amdgcn_s_wait_tensorcnt(0);
    __syncthreads();                       // tile `it` visible to all waves
    if (wv == 0 && it + 1 < nk) {          // prefetch tile it+1 into other buf
      const int ko = (it + 1) * BK;
      tdm_load_2d((unsigned)(size_t)&As[cur ^ 1][0][0],
                  A + (size_t)m0 * K + ko, (unsigned)K, (unsigned)M, BK, BM,
                  (unsigned long long)K);
      tdm_load_2d((unsigned)(size_t)&Bs[cur ^ 1][0][0],
                  Bt + (size_t)n0 * K + ko, (unsigned)K, (unsigned)Nn, BK, BN,
                  (unsigned long long)K);
    }
    compute_tile(As[cur], Bs[cur], wm, wn, half, r, acc);
    __syncthreads();                       // done reading before buf is reused
    cur ^= 1;
  }
#else
  for (int it = 0; it < nk; ++it) {
    const int ko = it * BK;
    __syncthreads();
#pragma unroll
    for (int p = 0; p < 2; ++p) {          // stage 128x32 A and B tiles
      int idx = p * 256 + tid;
      int row = idx >> 2;
      int kk  = (idx & 3) << 3;
      *(uint4*)(&As[0][row][kk]) = *(const uint4*)(A + (size_t)(m0 + row) * K + ko + kk);
      *(uint4*)(&Bs[0][row][kk]) = *(const uint4*)(Bt + (size_t)(n0 + row) * K + ko + kk);
    }
    __syncthreads();
    compute_tile(As[0], Bs[0], wm, wn, half, r, acc);
  }
#endif

  // ---- epilogue: C/D layout lane -> (m = vgpr + 8*half, n = r) ----
#pragma unroll
  for (int mi = 0; mi < 2; ++mi)
#pragma unroll
    for (int ni = 0; ni < 4; ++ni)
#pragma unroll
      for (int vv = 0; vv < 8; ++vv) {
        int m = m0 + wm * 32 + mi * 16 + half * 8 + vv;
        int n = n0 + wn * 64 + ni * 16 + r;
        float val = acc[mi][ni][vv] + bias[n];
        if (mode == 2) {
          Cres[(size_t)m * Nn + n] += val;
        } else {
          if (mode == 1)
            val = 0.5f * val * (1.0f + erff(val * 0.70710678118654752f));
          Cbf[(size_t)m * Nn + n] = (bf16_t)val;
        }
      }
}

// ---------------------------------------------------------------------------
// Banded self-attention (semantics match reference exactly in fp32):
//  - mask=1 && t>0: only keys {0} U [t-7,t+7] survive (out-of-band exp
//    underflows to 0) -> <=16-key wave path.
//  - mask=0 or t==0: additive mask is uniform over k -> softmax of raw
//    scores over all 512 keys.
// One block per (t, b); 12 waves = 12 heads; fused residual add.
// ---------------------------------------------------------------------------
__global__ __launch_bounds__(384)
void attn_kernel(const bf16_t* __restrict__ qb, const bf16_t* __restrict__ kb,
                 const bf16_t* __restrict__ vb, const int* __restrict__ mask,
                 float* __restrict__ xres) {
  const int t = blockIdx.x;
  const int b = blockIdx.y;
  const int h = threadIdx.x >> 5;
  const int lane = threadIdx.x & 31;

  __shared__ float probs[kH][kT];
  __shared__ float qsh[kH][kDH];
  __shared__ float psh[kH][16];
  __shared__ int   kidx[kH][16];

  const size_t rowq = ((size_t)b * kT + t) * kD + h * kDH;
  qsh[h][lane]      = (float)qb[rowq + lane];
  qsh[h][lane + 32] = (float)qb[rowq + lane + 32];
  __builtin_amdgcn_wave_barrier();

  const int   mk    = mask[b * kT + t];
  const float scale = 0.125f;  // 1/sqrt(64)
  const size_t base = ((size_t)b * kT) * kD + h * kDH;

  if (mk != 0 && t != 0) {
    int lo = t - 7; if (lo < 0) lo = 0;
    int hi = t + 7; if (hi > kT - 1) hi = kT - 1;
    int extra = (lo > 0) ? 1 : 0;
    int cnt = hi - lo + 1 + extra;  // <= 16

    float s = -3.0e38f;
    int key = -1;
    if (lane < cnt) {
      key = (extra && lane == 0) ? 0 : lo + lane - extra;
      const bf16_t* kp = kb + base + (size_t)key * kD;
      float d = 0.f;
#pragma unroll 8
      for (int j = 0; j < kDH; ++j) d += qsh[h][j] * (float)kp[j];
      s = d * scale;
    }
    float mx = s;
#pragma unroll
    for (int off = 16; off; off >>= 1) {
      float o = __shfl_xor(mx, off, 32);
      mx = mx > o ? mx : o;
    }
    float e = (lane < cnt) ? expf(s - mx) : 0.f;
    float sum = e;
#pragma unroll
    for (int off = 16; off; off >>= 1) sum += __shfl_xor(sum, off, 32);
    if (lane < 16) {
      psh[h][lane]  = (lane < cnt) ? (e / sum) : 0.f;
      kidx[h][lane] = (lane < cnt) ? key : 0;
    }
    __builtin_amdgcn_wave_barrier();
#pragma unroll
    for (int dd = 0; dd < 2; ++dd) {
      int d = lane + dd * 32;
      float c = 0.f;
      for (int j = 0; j < cnt; ++j)
        c += psh[h][j] * (float)vb[base + (size_t)kidx[h][j] * kD + d];
      xres[rowq + d] += c;
    }
  } else {
    for (int j = 0; j < 16; ++j) {
      int key = lane + j * 32;
      const bf16_t* kp = kb + base + (size_t)key * kD;
      float d = 0.f;
#pragma unroll 8
      for (int e2 = 0; e2 < kDH; ++e2) d += qsh[h][e2] * (float)kp[e2];
      probs[h][key] = d * scale;
    }
    __builtin_amdgcn_wave_barrier();
    float mx = -3.0e38f;
    for (int j = 0; j < 16; ++j) {
      float pv = probs[h][lane + j * 32];
      mx = mx > pv ? mx : pv;
    }
#pragma unroll
    for (int off = 16; off; off >>= 1) {
      float o = __shfl_xor(mx, off, 32);
      mx = mx > o ? mx : o;
    }
    float sum = 0.f;
    for (int j = 0; j < 16; ++j) {
      int key = lane + j * 32;
      float e = expf(probs[h][key] - mx);
      probs[h][key] = e;
      sum += e;
    }
#pragma unroll
    for (int off = 16; off; off >>= 1) sum += __shfl_xor(sum, off, 32);
    float inv = 1.f / sum;
    for (int j = 0; j < 16; ++j) probs[h][lane + j * 32] *= inv;
    __builtin_amdgcn_wave_barrier();
#pragma unroll
    for (int dd = 0; dd < 2; ++dd) {
      int d = lane + dd * 32;
      float c = 0.f;
      for (int key = 0; key < kT; ++key)
        c += probs[h][key] * (float)vb[base + (size_t)key * kD + d];
      xres[rowq + d] += c;
    }
  }
}

// ---------------------------------------------------------------------------
// Attention pooling + classifier epilogue
// ---------------------------------------------------------------------------
__global__ __launch_bounds__(256)
void pool_kernel(const float* __restrict__ x, const int* __restrict__ mask,
                 const float* __restrict__ pw, const float* __restrict__ pb,
                 float* __restrict__ pooled) {
  const int b = blockIdx.x, tid = threadIdx.x;
  __shared__ float wsc[kT];
  __shared__ float red[256];
  const float inv_sqrt_d = 0.03608439182435161f;  // 1/sqrt(768)

  for (int t = tid; t < kT; t += 256) {
    const float* xr = x + ((size_t)b * kT + t) * kD;
    float d = pb[0];
    for (int j = 0; j < kD; ++j) d += xr[j] * pw[j];
    d *= inv_sqrt_d;
    if (mask[b * kT + t] == 0) d += -10000.0f;
    wsc[t] = d;
  }
  __syncthreads();
  float m = fmaxf(wsc[tid], wsc[tid + 256]);
  red[tid] = m;
  __syncthreads();
  for (int s = 128; s; s >>= 1) {
    if (tid < s) red[tid] = fmaxf(red[tid], red[tid + s]);
    __syncthreads();
  }
  const float mx = red[0];
  __syncthreads();
  float e0 = expf(wsc[tid] - mx), e1 = expf(wsc[tid + 256] - mx);
  wsc[tid] = e0;
  wsc[tid + 256] = e1;
  red[tid] = e0 + e1;
  __syncthreads();
  for (int s = 128; s; s >>= 1) {
    if (tid < s) red[tid] += red[tid + s];
    __syncthreads();
  }
  const float inv_sum = 1.f / red[0];
  __syncthreads();
  wsc[tid] *= inv_sum;
  wsc[tid + 256] *= inv_sum;
  __syncthreads();
  for (int d = tid; d < kD; d += 256) {
    float acc = 0.f;
    for (int t = 0; t < kT; ++t)
      acc += x[((size_t)b * kT + t) * kD + d] * wsc[t];
    pooled[(size_t)b * kD + d] = acc;
  }
}

__global__ __launch_bounds__(256)
void fc_kernel(const float* __restrict__ pooled, const float* __restrict__ fw,
               const float* __restrict__ fb, float* __restrict__ out) {
  const int b = blockIdx.x, c = threadIdx.x;
  const float* p = pooled + (size_t)b * kD;
  float acc = fb[c];
  for (int d = 0; d < kD; ++d) acc += p[d] * fw[(size_t)d * kC + c];
  out[(size_t)b * kC + c] = acc;
}

// ---------------------------------------------------------------------------
// Launcher
// ---------------------------------------------------------------------------
extern "C" void kernel_launch(void* const* d_in, const int* in_sizes, int n_in,
                              void* d_out, int out_size, void* d_ws, size_t ws_size,
                              hipStream_t stream) {
  (void)in_sizes; (void)n_in; (void)out_size; (void)ws_size;
  const float* x      = (const float*)d_in[0];
  const int*   mask   = (const int*)d_in[1];
  const float* qkv_w  = (const float*)d_in[2];
  const float* qkv_b  = (const float*)d_in[3];
  const float* ln1_s  = (const float*)d_in[4];
  const float* ln1_b  = (const float*)d_in[5];
  const float* ln2_s  = (const float*)d_in[6];
  const float* ln2_b  = (const float*)d_in[7];
  const float* ffn_w1 = (const float*)d_in[8];
  const float* ffn_b1 = (const float*)d_in[9];
  const float* ffn_w2 = (const float*)d_in[10];
  const float* ffn_b2 = (const float*)d_in[11];
  const float* pool_w = (const float*)d_in[12];
  const float* pool_b = (const float*)d_in[13];
  const float* fc_w   = (const float*)d_in[14];
  const float* fc_b   = (const float*)d_in[15];
  float* out = (float*)d_out;

  char* ws = (char*)d_ws;
  size_t off = 0;
  auto take = [&](size_t bytes) -> char* {
    char* p = ws + off;
    off += (bytes + 255) & ~(size_t)255;
    return p;
  };

  float*  xbuf = (float*)take((size_t)kM * kD * 4);   // residual stream
  bf16_t* ybuf = (bf16_t*)take((size_t)kM * kD * 2);  // LN output
  bf16_t* qbuf = (bf16_t*)take((size_t)kM * kD * 2);
  bf16_t* kbuf = (bf16_t*)take((size_t)kM * kD * 2);
  bf16_t* vbuf = (bf16_t*)take((size_t)kM * kD * 2);
  bf16_t* hbuf = (bf16_t*)take((size_t)kM * kFF * 2); // FFN hidden (post-GELU)
  bf16_t* wqkv = (bf16_t*)take((size_t)kL * 3 * kD * kD * 2);   // transposed
  bf16_t* wf1  = (bf16_t*)take((size_t)kL * kD * kFF * 2);      // transposed
  bf16_t* wf2  = (bf16_t*)take((size_t)kL * kFF * kD * 2);      // transposed
  float*  pooled = (float*)take((size_t)kN * kD * 4);

  hipMemcpyAsync(xbuf, x, (size_t)kM * kD * 4, hipMemcpyDeviceToDevice, stream);

  // Weight transpose + bf16 conversion: B operands become [Nn][K] row-major.
  const dim3 tb(32, 8);
  cvtT_kernel<<<dim3(kD / 32, kD / 32, kL * 3), tb, 0, stream>>>(qkv_w, wqkv, kD, kD);
  cvtT_kernel<<<dim3(kFF / 32, kD / 32, kL), tb, 0, stream>>>(ffn_w1, wf1, kD, kFF);
  cvtT_kernel<<<dim3(kD / 32, kFF / 32, kL), tb, 0, stream>>>(ffn_w2, wf2, kFF, kD);

  const dim3 gD(kD / BN, kM / BM);     // 6 x 128
  const dim3 gFF(kFF / BN, kM / BM);   // 24 x 128

  for (int l = 0; l < kL; ++l) {
    ln_kernel<<<kM, 256, 0, stream>>>(xbuf, ln1_s + l * kD, ln1_b + l * kD, ybuf);
    for (int i = 0; i < 3; ++i) {
      bf16_t* ob = (i == 0) ? qbuf : ((i == 1) ? kbuf : vbuf);
      gemm_bf16_kernel<<<gD, 256, 0, stream>>>(
          ybuf, wqkv + (size_t)(l * 3 + i) * kD * kD,
          qkv_b + (size_t)(l * 3 + i) * kD, ob, nullptr, kM, kD, kD, 0);
    }
    attn_kernel<<<dim3(kT, kN), 384, 0, stream>>>(qbuf, kbuf, vbuf, mask, xbuf);

    ln_kernel<<<kM, 256, 0, stream>>>(xbuf, ln2_s + l * kD, ln2_b + l * kD, ybuf);
    gemm_bf16_kernel<<<gFF, 256, 0, stream>>>(
        ybuf, wf1 + (size_t)l * kD * kFF, ffn_b1 + (size_t)l * kFF,
        hbuf, nullptr, kM, kFF, kD, 1);
    gemm_bf16_kernel<<<gD, 256, 0, stream>>>(
        hbuf, wf2 + (size_t)l * kFF * kD, ffn_b2 + (size_t)l * kD,
        nullptr, xbuf, kM, kD, kFF, 2);
  }

  pool_kernel<<<kN, 256, 0, stream>>>(xbuf, mask, pool_w, pool_b, pooled);
  fc_kernel<<<kN, kC, 0, stream>>>(pooled, fc_w, fc_b, out);
}
